// StatePredictor_64716567216767
// MI455X (gfx1250) — matrix operand order repaired and checked by reading, compiled, not run on hardware
//
#include <hip/hip_runtime.h>
#include <math.h>

// ---------------- problem constants ----------------
#define Bsz   128
#define Lsz   256
#define Dsz   256
#define DIsz  512
#define DSsz  64
#define DTRsz 16
#define NLsz  4
#define DCONVs 4
#define HIDsz 256
#define EPSf  1e-5f
#define NXP   144              // DTR + 2*DS
#define NXPP  192              // NXP padded to multiple of TN

// GEMM tiling
#define TM 128
#define TN 64
#define TK 32
#define LDSS 40   // LDS row stride in shorts (80B: 16B-aligned b128 frags, bank spread)

typedef __attribute__((ext_vector_type(16))) __bf16 v16bf;
typedef __attribute__((ext_vector_type(8)))  float  v8f;

// ---------------- helpers ----------------
// Pack 4 f32 -> 4 bf16 (truncation) using v_perm_b32: one instruction per pair.
__device__ __forceinline__ unsigned long long pack4bf(float4 v) {
    unsigned a = __builtin_bit_cast(unsigned, v.x);
    unsigned b = __builtin_bit_cast(unsigned, v.y);
    unsigned c = __builtin_bit_cast(unsigned, v.z);
    unsigned d = __builtin_bit_cast(unsigned, v.w);
    // perm(S0,S1,sel): sel bytes 0-3 pick S1 bytes, 4-7 pick S0 bytes.
    // 0x07060302 -> {S0.b3,S0.b2,S1.b3,S1.b2} = {S0>>16, S1>>16}
    unsigned lo = __builtin_amdgcn_perm(b, a, 0x07060302u);  // [15:0]=bf16(x), [31:16]=bf16(y)
    unsigned hi = __builtin_amdgcn_perm(d, c, 0x07060302u);  // [15:0]=bf16(z), [31:16]=bf16(w)
    return ((unsigned long long)hi << 32) | lo;
}
__device__ __forceinline__ float siluf(float x) {
    return x * (1.0f / (1.0f + __expf(-x)));
}

// =====================================================================
// WMMA GEMM:  C[M,N] = epi( A[M,K] @ W[N,K]^T + bias[N] )
// block = 256 threads (8 waves).  Tile: 128(M) x 64(N), K-step 32.
// PRECONDITIONS (guaranteed by host):
//   - gridDim.y * TM <= M rows exist in A  (M in {32768, 128})
//   - K % 32 == 0
//   - W has >= ceil(N/TN)*TN rows (zero-padded where needed)
// Only the C-store is guarded (n < N, for N=144).
// Staging: float4 global loads -> v_perm bf16 pack -> ds_store_b64.
// epi: 0=none, 1=+bias, 2=+bias then silu
// =====================================================================
__global__ void wmma_gemm_kernel(const float* __restrict__ A, int lda,
                                 const float* __restrict__ W, int ldw,
                                 float* __restrict__ C, int ldc,
                                 int M, int N, int K,
                                 const float* __restrict__ bias, int epi)
{
    __shared__ unsigned short As[TM][LDSS];   // [m][k] bf16 bits
    __shared__ unsigned short Ws[TN][LDSS];   // [n][k] bf16 bits

    const int tid  = threadIdx.x;
    const int wave = tid >> 5;
    const int lane = tid & 31;
    const int mBase = blockIdx.y * TM;
    const int nBase = blockIdx.x * TN;

    // per-thread staging coordinates
    const int sr = tid >> 3;            // 0..31 (row group step 32), 8 float4 per row
    const int sk = (tid & 7) << 2;      // k quad

    v8f acc[4] = {};   // 4 x (16x16 f32) accumulators -> 16 x 64 per wave

    for (int k0 = 0; k0 < K; k0 += TK) {
        // ---- stage A tile: 128x32 f32 = 1024 float4 ; 4 per thread ----
#pragma unroll
        for (int i = 0; i < 4; ++i) {
            int r = sr + i * 32;
            float4 v = *(const float4*)(A + (long)(mBase + r) * lda + k0 + sk);
            *(unsigned long long*)&As[r][sk] = pack4bf(v);
        }
        // ---- stage W tile: 64x32 f32 = 512 float4 ; 2 per thread ----
#pragma unroll
        for (int i = 0; i < 2; ++i) {
            int r = sr + i * 32;
            float4 v = *(const float4*)(W + (long)(nBase + r) * ldw + k0 + sk);
            *(unsigned long long*)&Ws[r][sk] = pack4bf(v);
        }
        // prefetch next K tile (global_prefetch_b8)
        if (k0 + TK < K) {
            __builtin_prefetch(A + (long)(mBase + sr) * lda + k0 + TK + sk, 0, 1);
            __builtin_prefetch(W + (long)(nBase + sr) * ldw + k0 + TK + sk, 0, 1);
        }
        __syncthreads();

        // -------- A fragment (16x32 bf16, wave32 layout) --------
        // lanes 0-15: row=lane, K {0..7,16..23}; lanes 16-31: row=lane-16, K {8..15,24..31}
        const int ar  = lane & 15;
        const int akb = (lane >> 4) * 8;
        v16bf afrag;
#pragma unroll
        for (int e = 0; e < 16; ++e) {
            int k = (e < 8) ? (akb + e) : (16 + akb + (e - 8));
            afrag[e] = __builtin_bit_cast(__bf16, As[wave * 16 + ar][k]);
        }

        // -------- B fragments (32x16 bf16): lane = column, halves split K --------
        const int bn  = lane & 15;
    const int bkb = (lane >> 4) * 8;
#pragma unroll
        for (int nt = 0; nt < 4; ++nt) {
            v16bf bfrag;
#pragma unroll
            for (int e = 0; e < 16; ++e)
                bfrag[e] = __builtin_bit_cast(__bf16, Ws[nt * 16 + bn][(e < 8) ? (bkb + e) : (16 + bkb + (e - 8))]);
            acc[nt] = __builtin_amdgcn_wmma_f32_16x16x32_bf16(
                false, afrag, false, bfrag, (short)0, acc[nt], false, false);
        }
        __syncthreads();
    }

    // -------- epilogue: C layout VGPR r -> M = r + 8*(lane>=16), N = lane&15 --------
    const int cr = (lane >> 4) * 8;
    const int cn = lane & 15;
#pragma unroll
    for (int nt = 0; nt < 4; ++nt) {
        int n = nBase + nt * 16 + cn;
        if (n < N) {
            float bv = (epi >= 1) ? bias[n] : 0.0f;
#pragma unroll
            for (int r = 0; r < 8; ++r) {
                int m = mBase + wave * 16 + cr + r;
                float v = acc[nt][r] + bv;
                if (epi == 2) v = siluf(v);
                C[(long)m * ldc + n] = v;
            }
        }
    }
}

// =====================================================================
// Zero-pad xproj weights: (NL,144,512) -> (NL,192,512)
// =====================================================================
__global__ void pad_xw_kernel(const float* __restrict__ xw,
                              float* __restrict__ xwp)
{
    long idx = (long)blockIdx.x * 256 + threadIdx.x;
    long total = (long)NLsz * NXPP * DIsz;
    if (idx >= total) return;
    int  k   = (int)(idx % DIsz);
    long r   = idx / DIsz;
    int  row = (int)(r % NXPP);
    int  lay = (int)(r / NXPP);
    xwp[idx] = (row < NXP) ? xw[((long)lay * NXP + row) * DIsz + k] : 0.0f;
}

// =====================================================================
// Build input sequence: HID[b,l,:] = (l<L-1 ? hist : z_proj) + pos_emb
// =====================================================================
__global__ void build_seq_kernel(const float* __restrict__ hist,
                                 const float* __restrict__ zp,
                                 const float* __restrict__ pos,
                                 float* __restrict__ hid)
{
    long idx = (long)blockIdx.x * 256 + threadIdx.x;
    if (idx >= (long)Bsz * Lsz * Dsz) return;
    int  dch = (int)(idx % Dsz);
    long bl  = idx / Dsz;
    int  l   = (int)(bl % Lsz);
    long b   = bl / Lsz;
    float v = (l < Lsz - 1) ? hist[(b * (Lsz - 1) + l) * Dsz + dch]
                            : zp[b * Dsz + dch];
    hid[idx] = v + pos[(long)l * Dsz + dch];
}

// =====================================================================
// Fused residual add + RMSNorm.  One row (D=256) per 256-thread block.
// =====================================================================
__global__ void add_rms_kernel(float* __restrict__ residual,
                               float* __restrict__ hidden,
                               const float* __restrict__ w, int doAdd)
{
    long row = blockIdx.x;
    long idx = row * Dsz + threadIdx.x;
    float v = hidden[idx];
    if (doAdd) v += residual[idx];
    residual[idx] = v;

    float s = v * v;
#pragma unroll
    for (int o = 16; o > 0; o >>= 1) s += __shfl_down(s, o, 32);
    __shared__ float red[8];
    __shared__ float tot;
    if ((threadIdx.x & 31) == 0) red[threadIdx.x >> 5] = s;
    __syncthreads();
    if (threadIdx.x == 0) {
        float t = 0.f;
#pragma unroll
        for (int i = 0; i < 8; ++i) t += red[i];
        tot = t;
    }
    __syncthreads();
    float scale = rsqrtf(tot / (float)Dsz + EPSf);
    hidden[idx] = v * scale * w[threadIdx.x];
}

// =====================================================================
// Depthwise causal conv (DCONV=4) + bias + silu.
// =====================================================================
__global__ void conv_silu_kernel(const float* __restrict__ xz,
                                 const float* __restrict__ cw,
                                 const float* __restrict__ cb,
                                 float* __restrict__ u)
{
    long idx = (long)blockIdx.x * 256 + threadIdx.x;
    if (idx >= (long)Bsz * Lsz * DIsz) return;
    int  c  = (int)(idx % DIsz);
    long bl = idx / DIsz;
    int  l  = (int)(bl % Lsz);
    long b  = bl / Lsz;
    float s = cb[c];
#pragma unroll
    for (int j = 0; j < DCONVs; ++j) {
        int ls = l - (DCONVs - 1) + j;
        if (ls >= 0) s += xz[((b * Lsz + ls) * (2 * DIsz)) + c] * cw[c * DCONVs + j];
    }
    u[idx] = siluf(s);
}

// =====================================================================
// Fused selective scan: delta GEMV(K=16)+softplus, h-recurrence (state in
// registers), y = (h.C + u*D)*silu(z), written in place over u.
// =====================================================================
__global__ void scan_kernel(const float* __restrict__ dbc,
                            float* __restrict__ u,          // in: u, out: y
                            const float* __restrict__ xz,   // z gate at [...,512+d]
                            const float* __restrict__ dtw,
                            const float* __restrict__ dtb,
                            const float* __restrict__ Alog,
                            const float* __restrict__ Dp)
{
    const int b   = blockIdx.y;
    const int d   = blockIdx.x * 256 + threadIdx.x;   // DI=512 -> gridDim.x=2
    const int tid = threadIdx.x;

    float negA[DSsz];
#pragma unroll
    for (int n = 0; n < DSsz; ++n) negA[n] = -__expf(Alog[(long)d * DSsz + n]);
    float wdt[DTRsz];
#pragma unroll
    for (int r = 0; r < DTRsz; ++r) wdt[r] = dtw[(long)d * DTRsz + r];
    const float bdt = dtb[d];
    const float Dd  = Dp[d];

    float h[DSsz];
#pragma unroll
    for (int n = 0; n < DSsz; ++n) h[n] = 0.0f;

    __shared__ float sDt[DTRsz], sB[DSsz], sC[DSsz];

    for (int t = 0; t < Lsz; ++t) {
        long row = ((long)b * Lsz + t) * NXP;   // 144
        if (tid < 16)       sDt[tid]      = dbc[row + tid];
        else if (tid < 80)  sB[tid - 16]  = dbc[row + tid];
        else if (tid < 144) sC[tid - 80]  = dbc[row + tid];
        __syncthreads();

        float dtv = bdt;
#pragma unroll
        for (int r = 0; r < DTRsz; ++r) dtv += sDt[r] * wdt[r];
        float delta = (dtv > 20.0f) ? dtv : __logf(1.0f + __expf(dtv));

        long uidx = ((long)b * Lsz + t) * DIsz + d;
        float ut = u[uidx];
        float du = delta * ut;

        float acc = 0.0f;
#pragma unroll
        for (int n = 0; n < DSsz; ++n) {
            h[n] = __expf(delta * negA[n]) * h[n] + du * sB[n];
            acc += h[n] * sC[n];
        }

        float z = xz[((long)b * Lsz + t) * (2 * DIsz) + DIsz + d];
        u[uidx] = (acc + ut * Dd) * siluf(z);
        __syncthreads();
    }
}

// =====================================================================
// GLU gate: g[row,h] = h1[row,h] * silu(h1[row,256+h])
// =====================================================================
__global__ void glu_gate_kernel(const float* __restrict__ h1,
                                float* __restrict__ g)
{
    long idx = (long)blockIdx.x * 256 + threadIdx.x;
    if (idx >= (long)Bsz * Lsz * HIDsz) return;
    int  hc  = (int)(idx % HIDsz);
    long row = idx / HIDsz;
    float a  = h1[row * (2 * HIDsz) + hc];
    float gg = h1[row * (2 * HIDsz) + HIDsz + hc];
    g[idx] = a * siluf(gg);
}

// =====================================================================
// Final add + RMSNorm for the last token only -> LAST[b,:]
// =====================================================================
__global__ void final_rms_last_kernel(const float* __restrict__ residual,
                                      const float* __restrict__ hidden,
                                      const float* __restrict__ w,
                                      float* __restrict__ last)
{
    int b = blockIdx.x;
    long idx = ((long)b * Lsz + (Lsz - 1)) * Dsz + threadIdx.x;
    float v = hidden[idx] + residual[idx];

    float s = v * v;
#pragma unroll
    for (int o = 16; o > 0; o >>= 1) s += __shfl_down(s, o, 32);
    __shared__ float red[8];
    __shared__ float tot;
    if ((threadIdx.x & 31) == 0) red[threadIdx.x >> 5] = s;
    __syncthreads();
    if (threadIdx.x == 0) {
        float t = 0.f;
#pragma unroll
        for (int i = 0; i < 8; ++i) t += red[i];
        tot = t;
    }
    __syncthreads();
    float scale = rsqrtf(tot / (float)Dsz + EPSf);
    last[(long)b * Dsz + threadIdx.x] = v * scale * w[threadIdx.x];
}

// =====================================================================
// Final: z = res_w*p + (1-res_w)*z_t ; LayerNorm(z)*g + b  -> out
// =====================================================================
__global__ void final_ln_kernel(const float* __restrict__ P,
                                const float* __restrict__ z_t,
                                const float* __restrict__ res_w,
                                const float* __restrict__ g,
                                const float* __restrict__ beta,
                                float* __restrict__ out)
{
    int b = blockIdx.x;
    int tid = threadIdx.x;
    long idx = (long)b * Dsz + tid;
    float rw = res_w[0];
    float z = rw * P[idx] + (1.0f - rw) * z_t[idx];

    __shared__ float red[8];
    __shared__ float smu, svar;
    float s = z;
#pragma unroll
    for (int o = 16; o > 0; o >>= 1) s += __shfl_down(s, o, 32);
    if ((tid & 31) == 0) red[tid >> 5] = s;
    __syncthreads();
    if (tid == 0) {
        float t = 0.f;
#pragma unroll
        for (int i = 0; i < 8; ++i) t += red[i];
        smu = t / (float)Dsz;
    }
    __syncthreads();
    float mu = smu;
    float dv = z - mu;
    float s2 = dv * dv;
#pragma unroll
    for (int o = 16; o > 0; o >>= 1) s2 += __shfl_down(s2, o, 32);
    if ((tid & 31) == 0) red[tid >> 5] = s2;
    __syncthreads();
    if (tid == 0) {
        float t = 0.f;
#pragma unroll
        for (int i = 0; i < 8; ++i) t += red[i];
        svar = t / (float)Dsz;
    }
    __syncthreads();
    out[idx] = dv * rsqrtf(svar + EPSf) * g[tid] + beta[tid];
}

// =====================================================================
// Host-side orchestration
// =====================================================================
static inline void launch_gemm(const float* A, int lda, const float* W, int ldw,
                               float* C, int ldc, int M, int N, int K,
                               const float* bias, int epi, hipStream_t stream)
{
    dim3 grid((N + TN - 1) / TN, (M + TM - 1) / TM);
    wmma_gemm_kernel<<<grid, 256, 0, stream>>>(A, lda, W, ldw, C, ldc, M, N, K, bias, epi);
}

extern "C" void kernel_launch(void* const* d_in, const int* in_sizes, int n_in,
                              void* d_out, int out_size, void* d_ws, size_t ws_size,
                              hipStream_t stream)
{
    const float* z_t   = (const float*)d_in[0];
    const float* hist  = (const float*)d_in[1];
    const float* inp_w = (const float*)d_in[2];
    const float* inp_b = (const float*)d_in[3];
    const float* pos   = (const float*)d_in[4];
    const float* n1w   = (const float*)d_in[5];
    const float* in_w  = (const float*)d_in[6];
    const float* cw    = (const float*)d_in[7];
    const float* cb    = (const float*)d_in[8];
    const float* xw    = (const float*)d_in[9];
    const float* dtw   = (const float*)d_in[10];
    const float* dtb   = (const float*)d_in[11];
    const float* Alog  = (const float*)d_in[12];
    const float* Dpar  = (const float*)d_in[13];
    const float* ow    = (const float*)d_in[14];
    const float* n2w   = (const float*)d_in[15];
    const float* fc1w  = (const float*)d_in[16];
    const float* fc2w  = (const float*)d_in[17];
    const float* nfw   = (const float*)d_in[18];
    const float* phw1  = (const float*)d_in[19];
    const float* phb1  = (const float*)d_in[20];
    const float* phw2  = (const float*)d_in[21];
    const float* phb2  = (const float*)d_in[22];
    const float* lng   = (const float*)d_in[23];
    const float* lnb   = (const float*)d_in[24];
    const float* resw  = (const float*)d_in[25];
    float* out = (float*)d_out;

    const long BL  = (long)Bsz * Lsz;          // 32768 rows
    const long MLD = BL * Dsz;                 // 8.39M floats

    float* ws   = (float*)d_ws;
    float* RES  = ws;                          // (B,L,D)
    float* HID  = ws + MLD;                    // (B,L,D) hidden / normed
    float* XZ   = ws + 2 * MLD;                // (B,L,2*DI) = 4*MLD ; aliased by H1
    float* Ubuf = ws + 6 * MLD;                // (B,L,DI) = 2*MLD  ; aliased by Y, gated
    float* DBC  = ws + 8 * MLD;                // (B,L,144)
    float* LAST = DBC + BL * NXP;
    float* P1   = LAST + (long)Bsz * Dsz;
    float* Pp   = P1 + (long)Bsz * 2 * Dsz;
    float* ZP   = Pp + (long)Bsz * Dsz;
    float* XWP  = ZP + (long)Bsz * Dsz;        // (NL,192,512) padded xproj weights

    // ---- pad xproj weights (removes all staging guards from GEMM) ----
    {
        long n = (long)NLsz * NXPP * DIsz;
        pad_xw_kernel<<<(unsigned)((n + 255) / 256), 256, 0, stream>>>(xw, XWP);
    }

    // ---- input projection + sequence build ----
    launch_gemm(z_t, Dsz, inp_w, Dsz, ZP, Dsz, Bsz, Dsz, Dsz, inp_b, 1, stream);
    {
        long n = MLD;
        build_seq_kernel<<<(unsigned)((n + 255) / 256), 256, 0, stream>>>(hist, ZP, pos, HID);
    }

    // ---- layer stack ----
    for (int i = 0; i < NLsz; ++i) {
        // residual (+=) + rms -> HID
        add_rms_kernel<<<(unsigned)BL, 256, 0, stream>>>(RES, HID, n1w + (long)i * Dsz, i > 0 ? 1 : 0);

        // xz = x @ in_w^T          (32768 x 1024 x 256)
        launch_gemm(HID, Dsz, in_w + (long)i * 2 * DIsz * Dsz, Dsz,
                    XZ, 2 * DIsz, (int)BL, 2 * DIsz, Dsz, nullptr, 0, stream);

        // causal depthwise conv + silu -> U
        {
            long n = BL * DIsz;
            conv_silu_kernel<<<(unsigned)((n + 255) / 256), 256, 0, stream>>>(
                XZ, cw + (long)i * DIsz * DCONVs, cb + (long)i * DIsz, Ubuf);
        }

        // dbc = u @ xproj^T        (32768 x 144 x 512), padded weights
        launch_gemm(Ubuf, DIsz, XWP + (long)i * NXPP * DIsz, DIsz,
                    DBC, NXP, (int)BL, NXP, DIsz, nullptr, 0, stream);

        // fused selective scan, y in place over U
        {
            dim3 grid(DIsz / 256, Bsz);
            scan_kernel<<<grid, 256, 0, stream>>>(DBC, Ubuf, XZ,
                                                  dtw + (long)i * DIsz * DTRsz,
                                                  dtb + (long)i * DIsz,
                                                  Alog + (long)i * DIsz * DSsz,
                                                  Dpar + (long)i * DIsz);
        }

        // mixer out = y @ ow^T -> HID   (32768 x 256 x 512)
        launch_gemm(Ubuf, DIsz, ow + (long)i * Dsz * DIsz, DIsz,
                    HID, Dsz, (int)BL, Dsz, DIsz, nullptr, 0, stream);

        // residual += ; rms -> HID
        add_rms_kernel<<<(unsigned)BL, 256, 0, stream>>>(RES, HID, n2w + (long)i * Dsz, 1);

        // h1 = x @ fc1^T -> XZ (alias)   (32768 x 512 x 256)
        launch_gemm(HID, Dsz, fc1w + (long)i * 2 * HIDsz * Dsz, Dsz,
                    XZ, 2 * HIDsz, (int)BL, 2 * HIDsz, Dsz, nullptr, 0, stream);

        // gated = a * silu(g) -> U (alias)
        {
            long n = BL * HIDsz;
            glu_gate_kernel<<<(unsigned)((n + 255) / 256), 256, 0, stream>>>(XZ, Ubuf);
        }

        // hidden = gated @ fc2^T -> HID  (32768 x 256 x 256)
        launch_gemm(Ubuf, HIDsz, fc2w + (long)i * Dsz * HIDsz, HIDsz,
                    HID, Dsz, (int)BL, Dsz, HIDsz, nullptr, 0, stream);
    }

    // ---- final norm on last token + prediction head ----
    final_rms_last_kernel<<<Bsz, 256, 0, stream>>>(RES, HID, nfw, LAST);

    launch_gemm(LAST, Dsz, phw1, Dsz, P1, 2 * Dsz, Bsz, 2 * Dsz, Dsz, phb1, 2, stream); // +bias, silu
    launch_gemm(P1, 2 * Dsz, phw2, 2 * Dsz, Pp, Dsz, Bsz, Dsz, 2 * Dsz, phb2, 1, stream);

    final_ln_kernel<<<Bsz, 256, 0, stream>>>(Pp, z_t, resw, lng, lnb, out);
}
